// MambaSSM_83580063580381
// MI455X (gfx1250) — compile-verified, matrix operand order checked by reading, not run on hardware
//
#include <hip/hip_runtime.h>
#include <hip/hip_bf16.h>
#include <cmath>

// ---------------------------------------------------------------------------
// MambaSSM on MI455X (gfx1250, wave32, WMMA)
//   BATCH=16, SEQ=2048, D_IN=D_H=1024
// Pipeline:
//   1) cvt_f32_f16  : A,B,C -> f16 copies in ws (weights L2-resident, 6 MB)
//   2) bu_gemm      : Bu = x @ B^T  (f16 WMMA, fp32 acc) -> stored into d_out
//   3) mamba_scan   : 16 resident blocks (one per WGP), column-sliced
//                     recurrence; per-block A/C weight slices (256 KB) are
//                     parked in LDS ONCE via global_load_async_to_lds_b128
//                     (320 KB/WGP on CDNA5); h broadcast via global + device
//                     barrier per step, staged into LDS asynchronously;
//                     y_t overwrites Bu in d_out in place
//   4) ln_kernel    : row-wise LayerNorm in place on d_out
// ---------------------------------------------------------------------------

typedef __attribute__((ext_vector_type(16))) _Float16 v16h_t;
typedef __attribute__((ext_vector_type(8)))  _Float16 v8h_t;
typedef __attribute__((ext_vector_type(4)))  _Float16 v4h_t;
typedef __attribute__((ext_vector_type(8)))  float    v8f_t;

#define SEQ_LEN   2048
#define DH        1024
#define NBATCH    16
#define SCAN_WGS  16
#define LN_EPS    1e-5f

// LDS layout for mamba_scan (dynamic shared memory):
//   [0,128K)    A-slice (64 rows x 1024 k, f16)
//   [128K,256K) C-slice
//   [256K,288K) h stage (16 x 1024 f16)
//   [288K,292K) pair-reduction buffer (4 x 32 x 8 f32)
#define SCAN_LDS_BYTES (292 * 1024)

// Load a 16x32 (rows x K) f16 fragment, rows have leading dim `ld`.
// CDNA5 16-bit A/B operand layout: lane holds row = lane&15,
// VGPR v holds k = (v>=4?16:0) + 8*(lane>=16) + 2*(v&3) + {0,1}
// => per lane: two contiguous 16-byte chunks at k0 and k0+16.
__device__ __forceinline__ v16h_t load_frag_k(const _Float16* __restrict__ base,
                                              int ld, int lane) {
  const int row = lane & 15;
  const int kh  = (lane >> 4) << 3;   // 0 or 8
  const _Float16* p = base + (size_t)row * ld + kh;
  v8h_t lo = *(const v8h_t*)(p);      // k = kh .. kh+7
  v8h_t hi = *(const v8h_t*)(p + 16); // k = 16+kh .. 16+kh+7
  v16h_t f;
#pragma unroll
  for (int i = 0; i < 8; ++i) { f[i] = lo[i]; f[i + 8] = hi[i]; }
  return f;
}

__device__ __forceinline__ float gelu_exact(float v) {
  return 0.5f * v * (1.0f + erff(v * 0.7071067811865476f));
}

// Async-copy nbytes (multiple of 4096) global -> LDS in 16B chunks across
// 256 threads. Tracked by ASYNCcnt; data bypasses VGPRs entirely.
__device__ __forceinline__ void async_copy_to_lds(const void* __restrict__ gsrc,
                                                  void* sdst, uint32_t nbytes,
                                                  int tid) {
  // Low 32 bits of a generic LDS pointer are the LDS byte offset (aperture rule).
  uint32_t laddr = (uint32_t)(uintptr_t)sdst + (uint32_t)tid * 16u;
  uint64_t gaddr = (uint64_t)(uintptr_t)gsrc + (uint64_t)tid * 16u;
  const uint32_t niter = nbytes >> 12;   // / (256 threads * 16 B)
  for (uint32_t i = 0; i < niter; ++i) {
    asm volatile("global_load_async_to_lds_b128 %0, %1, off"
                 :: "v"(laddr), "v"(gaddr) : "memory");
    laddr += 4096u;
    gaddr += 4096u;
  }
}

__device__ __forceinline__ void wait_async_lds() {
#if __has_builtin(__builtin_amdgcn_s_wait_asynccnt)
  __builtin_amdgcn_s_wait_asynccnt(0);
#else
  asm volatile("s_wait_asynccnt 0" ::: "memory");
#endif
}

// ---------------------------------------------------------------------------
__global__ __launch_bounds__(256)
void mamba_cvt_f32_f16(const float* __restrict__ src, _Float16* __restrict__ dst, int n4) {
  int i = blockIdx.x * 256 + threadIdx.x;
  if (i < n4) {
    float4 v = ((const float4*)src)[i];
    v4h_t h = { (_Float16)v.x, (_Float16)v.y, (_Float16)v.z, (_Float16)v.w };
    ((v4h_t*)dst)[i] = h;
  }
}

// ---------------------------------------------------------------------------
// Bu = x @ B^T.  Grid: (2048 M-tiles, 2 N-halves), 256 threads (8 waves).
// Each block stages its 16x1024 x-tile (f32->f16, float4-vectorized) in LDS
// once; each wave computes 4 N-tiles of 16 columns with a K=1024 WMMA chain.
__global__ __launch_bounds__(256)
void mamba_bu_gemm(const float* __restrict__ x, const _Float16* __restrict__ Bh,
                   float* __restrict__ out) {
  __shared__ __align__(16) _Float16 xs[NBATCH * DH];   // 32 KB
  const int tid  = threadIdx.x;
  const int wave = tid >> 5;
  const int lane = tid & 31;
  const int mt   = blockIdx.x;

  {
    const float4* xsrc = (const float4*)(x + (size_t)mt * NBATCH * DH);
#pragma unroll
    for (int i = tid; i < NBATCH * DH / 4; i += 256) {
      float4 v = xsrc[i];
      v4h_t h = { (_Float16)v.x, (_Float16)v.y, (_Float16)v.z, (_Float16)v.w };
      *(v4h_t*)(xs + i * 4) = h;
    }
  }
  __syncthreads();

#pragma unroll
  for (int s = 0; s < 4; ++s) {
    const int ntile = blockIdx.y * 32 + wave * 4 + s;
    const _Float16* W = Bh + (size_t)ntile * 16 * DH;
    v8f_t acc = {};
#pragma unroll 4
    for (int ks = 0; ks < 32; ++ks) {
      v16h_t af = load_frag_k(xs + ks * 32, DH, lane);
      v16h_t bf = load_frag_k(W + ks * 32, DH, lane);
      acc = __builtin_amdgcn_wmma_f32_16x16x32_f16(false, af, false, bf,
                                                   (short)0, acc, false, false);
    }
    const int n = ntile * 16 + (lane & 15);
#pragma unroll
    for (int r = 0; r < 8; ++r) {
      const int m  = r + ((lane >> 4) << 3);          // row within tile
      const size_t gr = (size_t)mt * 16 + m;          // flat (b*SEQ + t)
      out[gr * DH + n] = acc[r];
    }
  }
}

// ---------------------------------------------------------------------------
// Sequential scan. 16 blocks x 256 threads; block owns 64 output columns
// (4 N-tiles); wave pair splits K (512 each), reduced through LDS.
// Weight slices live in LDS for the whole kernel (292 KB of the WGP's 320 KB).
__global__ __launch_bounds__(256)
void mamba_scan(const _Float16* __restrict__ Ah, const _Float16* __restrict__ Ch,
                float* __restrict__ inout,            // Bu in / y out, [b][t][h]
                _Float16* __restrict__ hbuf,          // 2 * 16*1024 f16, zeroed
                int* __restrict__ bar) {              // zeroed
  extern __shared__ __align__(16) char smem[];
  _Float16* Was = (_Float16*)smem;                         // 128 KB
  _Float16* Wcs = (_Float16*)(smem + 128 * 1024);          // 128 KB
  _Float16* hs  = (_Float16*)(smem + 256 * 1024);          //  32 KB
  float (*red)[32][8] = (float (*)[32][8])(smem + 288 * 1024); // 4 KB

  const int tid  = threadIdx.x;
  const int wave = tid >> 5;
  const int lane = tid & 31;
  const int nt   = wave >> 1;                 // n-tile 0..3 in block
  const int kh   = wave & 1;                  // K-half 0/1
  const int col0 = blockIdx.x * 64 + nt * 16;
  const int n    = col0 + (lane & 15);

  // Park this block's contiguous 64-row weight slices in LDS (once).
  async_copy_to_lds(Ah + (size_t)blockIdx.x * 64 * DH, Was, 64 * DH * 2, tid);
  async_copy_to_lds(Ch + (size_t)blockIdx.x * 64 * DH, Wcs, 64 * DH * 2, tid);
  // Stage h_0 (zeros); thereafter hs is refreshed mid-step.
  async_copy_to_lds(hbuf, hs, NBATCH * DH * 2, tid);
  wait_async_lds();
  __syncthreads();

  const _Float16* Wa = Was + (size_t)(nt * 16) * DH + kh * 512;
  const _Float16* Wc = Wcs + (size_t)(nt * 16) * DH + kh * 512;
  const _Float16* hk = hs + kh * 512;

  int p = 0;
  for (int t = 0; t < SEQ_LEN; ++t) {
    // ---- h_new slice = gelu(h @ A^T + Bu_t) ------------------------------
    v8f_t acc = {};
#pragma unroll 4
    for (int ks = 0; ks < 16; ++ks) {
      v16h_t af = load_frag_k(hk + ks * 32, DH, lane);
      v16h_t bf = load_frag_k(Wa + ks * 32, DH, lane);
      acc = __builtin_amdgcn_wmma_f32_16x16x32_f16(false, af, false, bf,
                                                   (short)0, acc, false, false);
    }
    if (kh) {
#pragma unroll
      for (int r = 0; r < 8; ++r) red[nt][lane][r] = acc[r];
    }
    __syncthreads();
    if (!kh) {
      _Float16* hout = hbuf + (size_t)(p ^ 1) * NBATCH * DH;
#pragma unroll
      for (int r = 0; r < 8; ++r) {
        const int m = r + ((lane >> 4) << 3);     // batch index
        float v = acc[r] + red[nt][lane][r]
                + inout[((size_t)m * SEQ_LEN + t) * DH + n];
        hout[m * DH + n] = (_Float16)gelu_exact(v);
      }
    }

    // ---- device-wide barrier (release h_new, acquire everyone's slice) ---
    __threadfence();
    __syncthreads();
    if (tid == 0) {
      __hip_atomic_fetch_add(bar, 1, __ATOMIC_RELEASE, __HIP_MEMORY_SCOPE_AGENT);
      const int target = SCAN_WGS * (t + 1);
      while (__hip_atomic_load(bar, __ATOMIC_ACQUIRE, __HIP_MEMORY_SCOPE_AGENT) < target)
        __builtin_amdgcn_s_sleep(1);
    }
    __syncthreads();
    __builtin_amdgcn_fence(__ATOMIC_ACQUIRE, "agent");

    // ---- refresh hs with full h_{t+1} (async, straight into LDS) ---------
    async_copy_to_lds(hbuf + (size_t)(p ^ 1) * NBATCH * DH, hs, NBATCH * DH * 2, tid);
    wait_async_lds();
    __syncthreads();

    // ---- y_t slice = h_new @ C^T  (overwrites Bu_t in place) -------------
    v8f_t acy = {};
#pragma unroll 4
    for (int ks = 0; ks < 16; ++ks) {
      v16h_t af = load_frag_k(hk + ks * 32, DH, lane);
      v16h_t bf = load_frag_k(Wc + ks * 32, DH, lane);
      acy = __builtin_amdgcn_wmma_f32_16x16x32_f16(false, af, false, bf,
                                                   (short)0, acy, false, false);
    }
    if (kh) {
#pragma unroll
      for (int r = 0; r < 8; ++r) red[nt][lane][r] = acy[r];
    }
    __syncthreads();
    if (!kh) {
#pragma unroll
      for (int r = 0; r < 8; ++r) {
        const int m = r + ((lane >> 4) << 3);
        inout[((size_t)m * SEQ_LEN + t) * DH + n] = acy[r] + red[nt][lane][r];
      }
    }
    __syncthreads();   // red + hs reused next iteration
    p ^= 1;
  }
}

// ---------------------------------------------------------------------------
__global__ __launch_bounds__(256)
void mamba_layernorm(float* __restrict__ out, const float* __restrict__ gamma,
                     const float* __restrict__ beta) {
  __shared__ float rs[256], rs2[256];
  const int tid = threadIdx.x;
  float* p = out + (size_t)blockIdx.x * DH;

  float s = 0.f, s2 = 0.f;
  for (int i = tid; i < DH; i += 256) { float v = p[i]; s += v; s2 += v * v; }
  rs[tid] = s; rs2[tid] = s2;
  __syncthreads();
  for (int off = 128; off > 0; off >>= 1) {
    if (tid < off) { rs[tid] += rs[tid + off]; rs2[tid] += rs2[tid + off]; }
    __syncthreads();
  }
  const float mean = rs[0] * (1.0f / DH);
  const float var  = rs2[0] * (1.0f / DH) - mean * mean;
  const float inv  = rsqrtf(var + LN_EPS);
  for (int i = tid; i < DH; i += 256)
    p[i] = (p[i] - mean) * inv * gamma[i] + beta[i];
}

// ---------------------------------------------------------------------------
extern "C" void kernel_launch(void* const* d_in, const int* in_sizes, int n_in,
                              void* d_out, int out_size, void* d_ws, size_t ws_size,
                              hipStream_t stream) {
  (void)in_sizes; (void)n_in; (void)out_size; (void)ws_size;

  const float* x     = (const float*)d_in[0];
  const float* A     = (const float*)d_in[1];
  const float* B     = (const float*)d_in[2];
  const float* C     = (const float*)d_in[3];
  const float* gamma = (const float*)d_in[4];
  const float* beta  = (const float*)d_in[5];
  float* out = (float*)d_out;

  char* ws = (char*)d_ws;
  const size_t MB = 1024 * 1024;
  _Float16* Ah   = (_Float16*)(ws + 0 * MB);
  _Float16* Ch   = (_Float16*)(ws + 2 * MB);
  _Float16* Bh   = (_Float16*)(ws + 4 * MB);
  _Float16* hbuf = (_Float16*)(ws + 6 * MB);
  int*      bar  = (int*)(ws + 6 * MB + 2 * NBATCH * DH * sizeof(_Float16));

  // Allow the scan kernel to use 292 KB of the WGP's 320 KB LDS.
  // (Host-side attribute set; not a stream operation, graph-capture safe.)
  static bool attr_done = false;
  if (!attr_done) {
    hipFuncSetAttribute((const void*)mamba_scan,
                        hipFuncAttributeMaxDynamicSharedMemorySize,
                        SCAN_LDS_BYTES);
    attr_done = true;
  }

  const int wn4 = DH * DH / 4;
  mamba_cvt_f32_f16<<<wn4 / 256, 256, 0, stream>>>(A, Ah, wn4);
  mamba_cvt_f32_f16<<<wn4 / 256, 256, 0, stream>>>(C, Ch, wn4);
  mamba_cvt_f32_f16<<<wn4 / 256, 256, 0, stream>>>(B, Bh, wn4);

  // zero h_0 (both buffers) + barrier counter
  hipMemsetAsync(hbuf, 0, 2 * NBATCH * DH * sizeof(_Float16) + 256, stream);

  mamba_bu_gemm<<<dim3(SEQ_LEN * NBATCH / 16, 2), 256, 0, stream>>>(x, Bh, out);
  mamba_scan<<<SCAN_WGS, 256, SCAN_LDS_BYTES, stream>>>(Ah, Ch, out, hbuf, bar);
  mamba_layernorm<<<NBATCH * SEQ_LEN, 256, 0, stream>>>(out, gamma, beta);
}
